// EncoderLayer_1314259993024
// MI455X (gfx1250) — matrix operand order repaired and checked by reading, not compiled, not run on hardware
//
#include <hip/hip_runtime.h>

typedef __attribute__((ext_vector_type(16))) __bf16 v16bf;
typedef __attribute__((ext_vector_type(8)))  float  v8f;
typedef __attribute__((ext_vector_type(4)))  int    i32x4;

constexpr int Bc = 4, Sc = 2048, Dc = 1024, Hc = 16, DFFc = 4096, DNc = 64;
constexpr int BSc = Bc * Sc;  // 8192 rows

// ---- CDNA5 async global->LDS path (guarded; falls back to load+store) ----
#if defined(__HIP_DEVICE_COMPILE__) &&                                   \
    __has_builtin(__builtin_amdgcn_global_load_async_to_lds_b128) &&     \
    __has_builtin(__builtin_amdgcn_s_wait_asynccnt)
#define USE_ASYNC_LDS 1
#else
#define USE_ASYNC_LDS 0
#endif

#define AS1 __attribute__((address_space(1)))
#define AS3 __attribute__((address_space(3)))

__device__ __forceinline__ void async_copy16(const void* g, void* l) {
#if USE_ASYNC_LDS
  __builtin_amdgcn_global_load_async_to_lds_b128((AS1 i32x4*)g, (AS3 i32x4*)l,
                                                 0, 0);
#endif
}
__device__ __forceinline__ void async_wait0() {
#if USE_ASYNC_LDS
  __builtin_amdgcn_s_wait_asynccnt(0);
#endif
}

// ---------------- helpers ----------------
__device__ __forceinline__ unsigned short f2bf(float f) {
  unsigned u = __float_as_uint(f);
  u += 0x7FFFu + ((u >> 16) & 1u);          // round-to-nearest-even
  return (unsigned short)(u >> 16);
}

__device__ __forceinline__ v8f wmma_bf16(v16bf a, v16bf b, v8f c) {
  return __builtin_amdgcn_wmma_f32_16x16x32_bf16(false, a, false, b, (short)0, c,
                                                 false, false);
}

__device__ __forceinline__ float redmax16(float x) {
  x = fmaxf(x, __shfl_xor(x, 1));
  x = fmaxf(x, __shfl_xor(x, 2));
  x = fmaxf(x, __shfl_xor(x, 4));
  x = fmaxf(x, __shfl_xor(x, 8));
  return x;
}
__device__ __forceinline__ float redsum16(float x) {
  x += __shfl_xor(x, 1);
  x += __shfl_xor(x, 2);
  x += __shfl_xor(x, 4);
  x += __shfl_xor(x, 8);
  return x;
}

// ---------------- fp32 -> bf16 convert ----------------
__global__ void cvt_kernel(const float* __restrict__ in,
                           unsigned short* __restrict__ out, size_t n) {
  size_t i = (size_t)blockIdx.x * blockDim.x + threadIdx.x;
  size_t stride = (size_t)gridDim.x * blockDim.x;
  for (; i < n; i += stride) out[i] = f2bf(in[i]);
}

// ---------------- generic bf16 WMMA GEMM core ----------------
// C[M,N] = A[M,K](bf16,row-major) x B[K,N](bf16,row-major), fp32 accum.
// Epilogue: +bias (per col), +residual (fp32), ReLU, store fp32 or bf16.
template <int BM, int BN, int WM, int WN>
__device__ void gemm_core(const unsigned short* __restrict__ A, int lda,
                          const unsigned short* __restrict__ Bm, int ldb,
                          const float* __restrict__ bias,
                          const float* __restrict__ res, int ldr,
                          float* __restrict__ outF,
                          unsigned short* __restrict__ outB, int ldo,
                          int col_off, int K, int relu, int blockM, int blockN) {
  static_assert(BM == 128, "BM must be 128 (256-thread loader)");
  constexpr int BK = 32, LSTR = 40;               // 40-element LDS row (pad 8)
  constexpr int WAVES_N = BN / WN;
  constexpr int RM = WM / 16, RN = WN / 16;
  constexpr int CHB = BN / 8;                     // bf16 per thread for B tile

  __shared__ unsigned short sA[BM * LSTR];        // [m][k]
  __shared__ unsigned short sB[BN * LSTR];        // [n][k] (transposed)

  const int tid = threadIdx.x;
  const int lane = tid & 31, wave = tid >> 5;
  const int wM = wave / WAVES_N, wN = wave % WAVES_N;
  const int half = lane >> 4, n16 = lane & 15;

  v8f acc[RM][RN];
#pragma unroll
  for (int i = 0; i < RM; i++)
#pragma unroll
    for (int j = 0; j < RN; j++)
#pragma unroll
      for (int v = 0; v < 8; v++) acc[i][j][v] = 0.0f;

  const int arow = tid >> 1, acol0 = (tid & 1) * 16;     // A: 128 rows x 32
  const int brow = tid >> 3, bcol0 = (tid & 7) * CHB;    // B: 32 rows x BN

  const unsigned short* Ag = A + (size_t)(blockM * BM + arow) * lda + acol0;
  const unsigned short* Bg0 = Bm + (size_t)blockN * BN + (size_t)brow * ldb + bcol0;

  for (int k0 = 0; k0 < K; k0 += BK) {
    // B tile -> registers (needs transpose into LDS)
    uint4 bld[CHB / 8];
#pragma unroll
    for (int i = 0; i < CHB / 8; i++)
      bld[i] = *(const uint4*)(Bg0 + (size_t)k0 * ldb + i * 8);

    if (k0 + BK < K) {
      __builtin_prefetch(Ag + k0 + BK, 0, 1);
      __builtin_prefetch(Bg0 + (size_t)(k0 + BK) * ldb, 0, 1);
    }

#if USE_ASYNC_LDS
    __syncthreads();  // previous iteration finished reading LDS
    // A tile: straight copy -> async DMA global->LDS (no VGPR round-trip)
    async_copy16(Ag + k0, &sA[arow * LSTR + acol0]);
    async_copy16(Ag + k0 + 8, &sA[arow * LSTR + acol0 + 8]);
    {
      const unsigned short* bs = (const unsigned short*)bld;
#pragma unroll
      for (int i = 0; i < CHB; i++) sB[(bcol0 + i) * LSTR + brow] = bs[i];
    }
    async_wait0();
    __syncthreads();
#else
    uint4 a0 = *(const uint4*)(Ag + k0);
    uint4 a1 = *(const uint4*)(Ag + k0 + 8);
    __syncthreads();
    *(uint4*)&sA[arow * LSTR + acol0] = a0;
    *(uint4*)&sA[arow * LSTR + acol0 + 8] = a1;
    {
      const unsigned short* bs = (const unsigned short*)bld;
#pragma unroll
      for (int i = 0; i < CHB; i++) sB[(bcol0 + i) * LSTR + brow] = bs[i];
    }
    __syncthreads();
#endif

    // A fragments: lane m = n16; VGPR0-3 -> K = 8*half + 0..7, VGPR4-7 -> +16
    v16bf af[RM];
#pragma unroll
    for (int i = 0; i < RM; i++) {
      int m = wM * WM + i * 16 + n16;
      uint4* u = reinterpret_cast<uint4*>(&af[i]);
      u[0] = *(const uint4*)&sA[m * LSTR + half * 8];
      u[1] = *(const uint4*)&sA[m * LSTR + half * 8 + 16];
    }
    // B fragments: lane n = n16; K = 8*half + 0..7, then +16 (same as A)
#pragma unroll
    for (int j = 0; j < RN; j++) {
      int n = wN * WN + j * 16 + n16;
      v16bf bf;
      uint4* u = reinterpret_cast<uint4*>(&bf);
      u[0] = *(const uint4*)&sB[n * LSTR + half * 8];
      u[1] = *(const uint4*)&sB[n * LSTR + half * 8 + 16];
#pragma unroll
      for (int i = 0; i < RM; i++) acc[i][j] = wmma_bf16(af[i], bf, acc[i][j]);
    }
  }

  const bool hasBias = bias != nullptr, hasRes = res != nullptr;
#pragma unroll
  for (int i = 0; i < RM; i++)
#pragma unroll
    for (int j = 0; j < RN; j++) {
      int col = blockN * BN + wN * WN + j * 16 + n16;
      int row0 = blockM * BM + wM * WM + i * 16 + half * 8;
      float bb = hasBias ? bias[col] : 0.0f;
#pragma unroll
      for (int v = 0; v < 8; v++) {
        int row = row0 + v;
        float xv = acc[i][j][v] + bb;
        if (hasRes) xv += res[(size_t)row * ldr + col];
        if (relu) xv = fmaxf(xv, 0.0f);
        size_t oidx = (size_t)row * ldo + col_off + col;
        if (outF) outF[oidx] = xv;
        else outB[oidx] = f2bf(xv);
      }
    }
}

template <int BM, int BN, int WM, int WN>
__global__ __launch_bounds__(256) void gemm_kernel(
    const unsigned short* __restrict__ A, int lda,
    const unsigned short* __restrict__ Bm, int ldb,
    const float* __restrict__ bias, const float* __restrict__ res, int ldr,
    float* __restrict__ outF, unsigned short* __restrict__ outB, int ldo,
    int K, int relu) {
  gemm_core<BM, BN, WM, WN>(A, lda, Bm, ldb, bias, res, ldr, outF, outB, ldo, 0,
                            K, relu, blockIdx.y, blockIdx.x);
}

// QKV: grid.z = 3*H, z -> (q/k/v, head). Output layout [B,S,H*DN] (col h*64+dn)
__global__ __launch_bounds__(256) void qkv_kernel(
    const unsigned short* __restrict__ xb, const unsigned short* __restrict__ wq,
    const unsigned short* __restrict__ wk, const unsigned short* __restrict__ wv,
    const float* __restrict__ bq, const float* __restrict__ bk,
    const float* __restrict__ bv, unsigned short* __restrict__ qb,
    unsigned short* __restrict__ kb, unsigned short* __restrict__ vb) {
  int z = blockIdx.z, which = z % 3, h = z / 3;
  const unsigned short* W =
      (which == 0 ? wq : which == 1 ? wk : wv) + (size_t)h * Dc * DNc;
  const float* bias = (which == 0 ? bq : which == 1 ? bk : bv) + h * DNc;
  unsigned short* out = which == 0 ? qb : which == 1 ? kb : vb;
  gemm_core<128, 64, 32, 32>(xb, Dc, W, DNc, bias, nullptr, 0, nullptr, out, Dc,
                             h * DNc, Dc, 0, blockIdx.y, 0);
}

// ---------------- flash attention (one wave per 16 query rows per (b,h)) ----
__global__ __launch_bounds__(32) void flash_kernel(
    const unsigned short* __restrict__ Q, const unsigned short* __restrict__ Kt,
    const unsigned short* __restrict__ Vt, const float* __restrict__ mask,
    unsigned short* __restrict__ Out) {
  constexpr int KSTR = 80;  // 64 dn + pad (16B aligned rows)
  constexpr int PSTR = 40;
  __shared__ unsigned short sk[32 * KSTR];
  __shared__ unsigned short sv[32 * KSTR];
  __shared__ unsigned short sp[16 * PSTR];

  const int lane = threadIdx.x;
  const int q0 = blockIdx.x * 16, h = blockIdx.y, b = blockIdx.z;
  const int half = lane >> 4, n16 = lane & 15;

  // Q A-fragments (dn 0..31, 32..63)
  v16bf qa[2];
  {
    const unsigned short* qp =
        Q + ((size_t)(b * Sc + q0 + n16)) * Dc + h * DNc;
#pragma unroll
    for (int s = 0; s < 2; s++) {
      int kbase = s * 32 + half * 8;
      uint4* u = reinterpret_cast<uint4*>(&qa[s]);
      u[0] = *(const uint4*)(qp + kbase);
      u[1] = *(const uint4*)(qp + kbase + 16);
    }
  }

  float m[8], l[8];
  v8f o[4];
#pragma unroll
  for (int v = 0; v < 8; v++) { m[v] = -1e30f; l[v] = 0.0f; }
#pragma unroll
  for (int c = 0; c < 4; c++)
#pragma unroll
    for (int v = 0; v < 8; v++) o[c][v] = 0.0f;

  const float* maskRow = mask + ((size_t)b * Sc + q0) * Sc;

  for (int kb = 0; kb < Sc / 32; kb++) {
    const int key0 = kb * 32;
    // stage K/V rows (lane -> one key row of 64 bf16)
    const unsigned short* kp = Kt + ((size_t)(b * Sc + key0 + lane)) * Dc + h * DNc;
    const unsigned short* vp = Vt + ((size_t)(b * Sc + key0 + lane)) * Dc + h * DNc;
#if USE_ASYNC_LDS
    __syncthreads();
#pragma unroll
    for (int i = 0; i < 8; i++) {
      async_copy16(kp + i * 8, &sk[lane * KSTR + i * 8]);
      async_copy16(vp + i * 8, &sv[lane * KSTR + i * 8]);
    }
    async_wait0();
    __syncthreads();
#else
    uint4 kr[8], vr[8];
#pragma unroll
    for (int i = 0; i < 8; i++) { kr[i] = ((const uint4*)kp)[i]; vr[i] = ((const uint4*)vp)[i]; }
    __syncthreads();
#pragma unroll
    for (int i = 0; i < 8; i++) {
      ((uint4*)&sk[lane * KSTR])[i] = kr[i];
      ((uint4*)&sv[lane * KSTR])[i] = vr[i];
    }
    __syncthreads();
#endif

    // scores tile 16x32 as two 16x16 accumulators
    v8f sc[2];
#pragma unroll
    for (int c = 0; c < 2; c++) {
      v8f z;
#pragma unroll
      for (int v = 0; v < 8; v++) z[v] = 0.0f;
      int krow = c * 16 + n16;
#pragma unroll
      for (int s = 0; s < 2; s++) {
        v16bf bf;
        uint4* u = reinterpret_cast<uint4*>(&bf);
        int kbx = s * 32 + half * 8;
        u[0] = *(const uint4*)&sk[krow * KSTR + kbx];
        u[1] = *(const uint4*)&sk[krow * KSTR + kbx + 16];
        z = wmma_bf16(qa[s], bf, z);
      }
      sc[c] = z;
    }

    // scale + mask
#pragma unroll
    for (int c = 0; c < 2; c++)
#pragma unroll
      for (int v = 0; v < 8; v++) {
        int row = v + half * 8;
        float s = sc[c][v] * 0.125f +
                  maskRow[(size_t)row * Sc + key0 + c * 16 + n16] * (-1e9f);
        sc[c][v] = s;
      }

    // online softmax
    float alpha[8];
#pragma unroll
    for (int v = 0; v < 8; v++) {
      float rmax = redmax16(fmaxf(sc[0][v], sc[1][v]));
      float mn = fmaxf(m[v], rmax);
      alpha[v] = __expf(m[v] - mn);
      m[v] = mn;
      float p0 = __expf(sc[0][v] - mn);
      float p1 = __expf(sc[1][v] - mn);
      sc[0][v] = p0; sc[1][v] = p1;
      float rs = redsum16(p0 + p1);
      l[v] = l[v] * alpha[v] + rs;
    }
#pragma unroll
    for (int c = 0; c < 4; c++)
#pragma unroll
      for (int v = 0; v < 8; v++) o[c][v] *= alpha[v];

    // P: C-layout -> A-layout via LDS
#pragma unroll
    for (int c = 0; c < 2; c++)
#pragma unroll
      for (int v = 0; v < 8; v++)
        sp[(v + half * 8) * PSTR + c * 16 + n16] = f2bf(sc[c][v]);
    __syncthreads();
    v16bf pf;
    {
      uint4* u = reinterpret_cast<uint4*>(&pf);
      u[0] = *(const uint4*)&sp[n16 * PSTR + half * 16];
      u[1] = *(const uint4*)&sp[n16 * PSTR + half * 16 + 8];
    }

    // O += P x V
#pragma unroll
    for (int c = 0; c < 4; c++) {
      v16bf vf;
      unsigned short* vu = (unsigned short*)&vf;
      int dn = c * 16 + n16;
#pragma unroll
      for (int v = 0; v < 8; v++) {
        int k2 = half * 16 + 2 * v;
        vu[2 * v] = sv[k2 * KSTR + dn];
        vu[2 * v + 1] = sv[(k2 + 1) * KSTR + dn];
      }
      o[c] = wmma_bf16(pf, vf, o[c]);
    }
  }

  // normalize and write concat (bf16, layout [B,S,H*DN])
#pragma unroll
  for (int v = 0; v < 8; v++) l[v] = 1.0f / l[v];
#pragma unroll
  for (int c = 0; c < 4; c++)
#pragma unroll
    for (int v = 0; v < 8; v++) {
      int row = q0 + v + half * 8;
      Out[((size_t)(b * Sc + row)) * Dc + h * DNc + c * 16 + n16] =
          f2bf(o[c][v] * l[v]);
    }
}

// ---------------- LayerNorm (one row per block, D=1024) ----------------
__global__ __launch_bounds__(256) void ln_kernel(
    const float* __restrict__ in, const float* __restrict__ g,
    const float* __restrict__ be, float* __restrict__ outF,
    unsigned short* __restrict__ outB) {
  const int row = blockIdx.x, tid = threadIdx.x;
  const float4 v = ((const float4*)(in + (size_t)row * Dc))[tid];
  float s = v.x + v.y + v.z + v.w;
  float q = v.x * v.x + v.y * v.y + v.z * v.z + v.w * v.w;
  for (int msk = 1; msk < 32; msk <<= 1) {
    s += __shfl_xor(s, msk);
    q += __shfl_xor(q, msk);
  }
  __shared__ float ws[8], wq2[8];
  __shared__ float smean, sinv;
  const int lane = tid & 31, wave = tid >> 5;
  if (lane == 0) { ws[wave] = s; wq2[wave] = q; }
  __syncthreads();
  if (tid == 0) {
    float ts = 0.0f, tq = 0.0f;
    for (int i = 0; i < 8; i++) { ts += ws[i]; tq += wq2[i]; }
    float mean = ts * (1.0f / Dc);
    float var = tq * (1.0f / Dc) - mean * mean;
    smean = mean;
    sinv = rsqrtf(var + 1e-5f);
  }
  __syncthreads();
  const float mean = smean, inv = sinv;
  const float4 gg = ((const float4*)g)[tid];
  const float4 bb = ((const float4*)be)[tid];
  float4 ov;
  ov.x = (v.x - mean) * inv * gg.x + bb.x;
  ov.y = (v.y - mean) * inv * gg.y + bb.y;
  ov.z = (v.z - mean) * inv * gg.z + bb.z;
  ov.w = (v.w - mean) * inv * gg.w + bb.w;
  ((float4*)(outF + (size_t)row * Dc))[tid] = ov;
  if (outB) {
    unsigned short* ob = outB + (size_t)row * Dc + tid * 4;
    ob[0] = f2bf(ov.x); ob[1] = f2bf(ov.y); ob[2] = f2bf(ov.z); ob[3] = f2bf(ov.w);
  }
}

// ---------------- host ----------------
extern "C" void kernel_launch(void* const* d_in, const int* in_sizes, int n_in,
                              void* d_out, int out_size, void* d_ws,
                              size_t ws_size, hipStream_t stream) {
  (void)in_sizes; (void)n_in; (void)out_size; (void)ws_size;
  const float* x = (const float*)d_in[0];
  const float* mask = (const float*)d_in[1];
  const float* Wq = (const float*)d_in[2];
  const float* bq = (const float*)d_in[3];
  const float* Wk = (const float*)d_in[4];
  const float* bk = (const float*)d_in[5];
  const float* Wv = (const float*)d_in[6];
  const float* bv = (const float*)d_in[7];
  const float* Wo = (const float*)d_in[8];
  const float* bo = (const float*)d_in[9];
  const float* W1 = (const float*)d_in[10];
  const float* b1 = (const float*)d_in[11];
  const float* W2 = (const float*)d_in[12];
  const float* b2 = (const float*)d_in[13];
  const float* g1 = (const float*)d_in[14];
  const float* be1 = (const float*)d_in[15];
  const float* g2 = (const float*)d_in[16];
  const float* be2 = (const float*)d_in[17];
  float* out = (float*)d_out;

  char* w = (char*)d_ws;
  size_t off = 0;
  auto alloc = [&](size_t bytes) -> void* {
    off = (off + 255) & ~(size_t)255;
    void* p = w + off;
    off += bytes;
    return p;
  };
  typedef unsigned short us;
  us* xb  = (us*)alloc((size_t)BSc * Dc * 2);
  us* wqb = (us*)alloc((size_t)Hc * Dc * DNc * 2);
  us* wkb = (us*)alloc((size_t)Hc * Dc * DNc * 2);
  us* wvb = (us*)alloc((size_t)Hc * Dc * DNc * 2);
  us* wob = (us*)alloc((size_t)Dc * Dc * 2);
  us* w1b = (us*)alloc((size_t)Dc * DFFc * 2);
  us* w2b = (us*)alloc((size_t)DFFc * Dc * 2);
  us* qb  = (us*)alloc((size_t)BSc * Dc * 2);
  us* kb  = (us*)alloc((size_t)BSc * Dc * 2);
  us* vb  = (us*)alloc((size_t)BSc * Dc * 2);
  us* cb  = (us*)alloc((size_t)BSc * Dc * 2);
  float* y1 = (float*)alloc((size_t)BSc * Dc * 4);
  float* x1 = (float*)alloc((size_t)BSc * Dc * 4);
  us* x1b = (us*)alloc((size_t)BSc * Dc * 2);
  us* hb  = (us*)alloc((size_t)BSc * DFFc * 2);
  float* y2 = (float*)alloc((size_t)BSc * Dc * 4);

  // fp32 -> bf16 conversions
  cvt_kernel<<<2048, 256, 0, stream>>>(x, xb, (size_t)BSc * Dc);
  cvt_kernel<<<1024, 256, 0, stream>>>(Wq, wqb, (size_t)Hc * Dc * DNc);
  cvt_kernel<<<1024, 256, 0, stream>>>(Wk, wkb, (size_t)Hc * Dc * DNc);
  cvt_kernel<<<1024, 256, 0, stream>>>(Wv, wvb, (size_t)Hc * Dc * DNc);
  cvt_kernel<<<1024, 256, 0, stream>>>(Wo, wob, (size_t)Dc * Dc);
  cvt_kernel<<<2048, 256, 0, stream>>>(W1, w1b, (size_t)Dc * DFFc);
  cvt_kernel<<<2048, 256, 0, stream>>>(W2, w2b, (size_t)DFFc * Dc);

  // QKV projections (48 GEMMs fused into one launch via grid.z)
  qkv_kernel<<<dim3(1, BSc / 128, Hc * 3), 256, 0, stream>>>(
      xb, wqb, wkb, wvb, bq, bk, bv, qb, kb, vb);

  // flash attention -> concat heads (bf16)
  flash_kernel<<<dim3(Sc / 16, Hc, Bc), 32, 0, stream>>>(qb, kb, vb, mask, cb);

  // output projection + bias + residual(x) -> y1 (fp32)
  gemm_kernel<128, 128, 64, 32><<<dim3(Dc / 128, BSc / 128), 256, 0, stream>>>(
      cb, Dc, wob, Dc, bo, x, Dc, y1, nullptr, Dc, Dc, 0);

  // LN1 -> x1 (fp32) + x1b (bf16)
  ln_kernel<<<BSc, 256, 0, stream>>>(y1, g1, be1, x1, x1b);

  // FFN1: relu(x1 @ W1 + b1) -> hb (bf16)
  gemm_kernel<128, 128, 64, 32><<<dim3(DFFc / 128, BSc / 128), 256, 0, stream>>>(
      x1b, Dc, w1b, DFFc, b1, nullptr, 0, nullptr, hb, DFFc, Dc, 1);

  // FFN2: hb @ W2 + b2 + residual(x1) -> y2 (fp32)
  gemm_kernel<128, 128, 64, 32><<<dim3(Dc / 128, BSc / 128), 256, 0, stream>>>(
      hb, DFFc, w2b, Dc, b2, x1, Dc, y2, nullptr, Dc, DFFc, 0);

  // LN2 -> out
  ln_kernel<<<BSc, 256, 0, stream>>>(y2, g2, be2, out, nullptr);
}
